// SpGraphAttentionLayer_2740189135492
// MI455X (gfx1250) — compile-verified, matrix operand order checked by reading, not run on hardware
//
#include <hip/hip_runtime.h>

typedef __attribute__((ext_vector_type(16))) _Float16 v16h;
typedef __attribute__((ext_vector_type(8)))  float    v8f;

// Load a 16-half WMMA fragment (A or B) from a contiguous K-major row in LDS.
// Per CDNA5 ISA 7.12.2 (16-bit A 16x32): lane half selects K sub-block,
// VGPR v holds K pair  k = (v&3)*2 + khalf*8 + (v>>2)*16  (+row base).
__device__ __forceinline__ v16h frag_ld(const _Float16* row, int khalf) {
  v16h f;
#pragma unroll
  for (int v = 0; v < 8; ++v) {
    const int k = ((v & 3) << 1) + (khalf << 3) + ((v >> 2) << 4);
    f[2 * v]     = row[k];
    f[2 * v + 1] = row[k + 1];
  }
  return f;
}

// CDNA5 hardware transcendental tanh (TRANS pipe, co-executes with WMMA).
__device__ __forceinline__ float fast_tanh(float x) {
  float r;
  asm volatile("v_tanh_f32 %0, %1" : "=v"(r) : "v"(x));
  return r;
}

// Float atomic max via ordered-int trick (init value must be -inf).
__device__ __forceinline__ void atomicMaxF32(float* addr, float val) {
  if (val >= 0.0f) atomicMax((int*)addr, __float_as_int(val));
  else             atomicMin((unsigned int*)addr, __float_as_uint(val));
}

// ------------------------------------------------ one-time weight pre-convert
// Pads + converts all GEMM weights to f16 in workspace so hot kernels do pure
// b128 copies into LDS (no per-block cvt, half the read bytes).
__global__ __launch_bounds__(256) void k_prep(
    const float* __restrict__ w_att_w, const float* __restrict__ w1,
    const float* __restrict__ w2, const float* __restrict__ w3,
    _Float16* __restrict__ wa16, _Float16* __restrict__ w1_16,
    _Float16* __restrict__ w2_16, _Float16* __restrict__ w3_16)
{
  const int i = blockIdx.x * 256 + threadIdx.x;
  if (i < 128 * 96) {   // w_att_w: (128,71) -> (128,96) padded
    const int n = i / 96, k = i - n * 96;
    wa16[i] = (k < 71) ? (_Float16)w_att_w[n * 71 + k] : (_Float16)0.0f;
  }
  if (i < 128 * 128) {
    const int n = i >> 7, k = i & 127;
    w1_16[i] = (k < 69) ? (_Float16)w1[n * 69 + k] : (_Float16)0.0f;
    w2_16[i] = (_Float16)w2[i];
    w3_16[i] = (_Float16)w3[i];
  }
}

// ---------------------------------------------------------------- init
__global__ __launch_bounds__(256) void k_init(float* h_att, float* mx, float* dn, int N) {
  const size_t i = (size_t)blockIdx.x * 256 + threadIdx.x;
  if (i < (size_t)N * 96) h_att[i] = 0.0f;
  if (i < (size_t)N) { mx[i] = -__builtin_inff(); dn[i] = 0.0f; }
}

// ---------------------------------------------------- edge attention logits
// Block: 256 threads = 8 waves, 128 edges (16 per wave).
// z[128] = xa(71, pad96) @ w_att_w^T ; att = sum_n va[n]*tanh(z[n] + b[n])
__global__ __launch_bounds__(256) void k_edge_att(
    const float* __restrict__ X_msg, const float* __restrict__ feature,
    const float* __restrict__ te_w, const float* __restrict__ te_lam,
    const float* __restrict__ shared_w, const float* __restrict__ shared_b,
    const _Float16* __restrict__ wa16, const float* __restrict__ w_att_b,
    const float* __restrict__ va,
    const int* __restrict__ src, const int* __restrict__ dst,
    const int* __restrict__ dst_lane,
    float* __restrict__ att, float* __restrict__ mx, int E)
{
  __shared__ alignas(16) _Float16 Wl[128 * 96];  // B matrix rows, K padded to 96
  __shared__ alignas(16) _Float16 xa[128 * 96];  // A rows, K padded to 96
  __shared__ float bl[128];
  __shared__ float vl[128];

  const int tid = threadIdx.x;
  // pure vector copy of pre-converted weights (1536 uint4)
  {
    const uint4* src16 = (const uint4*)wa16;
    uint4* dst16 = (uint4*)Wl;
    for (int i = tid; i < 128 * 96 / 8; i += 256) dst16[i] = src16[i];
  }
  if (tid < 128) { bl[tid] = w_att_b[tid]; vl[tid] = va[tid]; }

  // Build xa rows: 2 threads per edge.
  {
    const int row = tid >> 1, part = tid & 1;
    const int e = blockIdx.x * 128 + row;
    _Float16* xr = &xa[row * 96];
    if (e < E) {
      const float dt = feature[4 * e + 1];
      const int ln = dst_lane[e];
      const float rl = te_lam[ln];
      const float lam = __expf(-rl * rl);
      const float oml = 1.0f - lam;
      if (part == 0) {
        const int s = src[e], d = dst[e];
        xr[0] = (_Float16)X_msg[2 * s];
        xr[1] = (_Float16)X_msg[2 * s + 1];
        xr[2] = (_Float16)X_msg[2 * d];
        xr[3] = (_Float16)X_msg[2 * d + 1];
        xr[4] = (_Float16)feature[4 * e];
        xr[5] = (_Float16)dt;
        xr[6] = (_Float16)feature[4 * e + 2];
        for (int k = 71; k < 96; ++k) xr[k] = (_Float16)0.0f;
      }
      for (int d2 = part * 16; d2 < part * 16 + 16; ++d2) {
        const float ang = dt * te_w[ln * 32 + d2];
        const float sh  = dt * shared_w[d2] + shared_b[d2];
        xr[7 + d2]  = (_Float16)(oml * __sinf(ang) + lam * __sinf(sh));
        xr[39 + d2] = (_Float16)(oml * __cosf(ang) + lam * __cosf(sh));
      }
    } else {
      if (part == 0) {
        for (int k = 0; k < 7; ++k) xr[k] = (_Float16)0.0f;
        for (int k = 71; k < 96; ++k) xr[k] = (_Float16)0.0f;
      }
      for (int d2 = part * 16; d2 < part * 16 + 16; ++d2) {
        xr[7 + d2] = (_Float16)0.0f; xr[39 + d2] = (_Float16)0.0f;
      }
    }
  }
  __syncthreads();

  const int wave = tid >> 5, lane = tid & 31;
  const int khalf = lane >> 4, nl = lane & 15;

  // Zero-init accumulators -> first WMMA uses inline SRC2=0.
  v8f acc[8];
#pragma unroll
  for (int nt = 0; nt < 8; ++nt) acc[nt] = (v8f)(0.0f);

#pragma unroll
  for (int kt = 0; kt < 3; ++kt) {
    const v16h a = frag_ld(&xa[(wave * 16 + nl) * 96 + kt * 32], khalf);
    v16h bfr[8];                       // hoist: all 16 ds_load_b128 in flight
#pragma unroll
    for (int nt = 0; nt < 8; ++nt)
      bfr[nt] = frag_ld(&Wl[(nt * 16 + nl) * 96 + kt * 32], khalf);
#pragma unroll
    for (int nt = 0; nt < 8; ++nt)
      acc[nt] = __builtin_amdgcn_wmma_f32_16x16x32_f16(
          false, a, false, bfr[nt], (short)0, acc[nt], false, false);
  }

  // Epilogue: att[m] = sum_n va[n]*tanh(z[m][n]+b[n]); C layout:
  // VGPR r -> row r (lanes 0-15) / row r+8 (lanes 16-31), col n = nt*16+(lane&15).
  float bv[8], vv[8];
#pragma unroll
  for (int nt = 0; nt < 8; ++nt) { bv[nt] = bl[nt * 16 + nl]; vv[nt] = vl[nt * 16 + nl]; }

#pragma unroll
  for (int r = 0; r < 8; ++r) {
    float s = 0.0f;
#pragma unroll
    for (int nt = 0; nt < 8; ++nt)
      s = fmaf(vv[nt], fast_tanh(acc[nt][r] + bv[nt]), s);
    s += __shfl_xor(s, 1, 32);
    s += __shfl_xor(s, 2, 32);
    s += __shfl_xor(s, 4, 32);
    s += __shfl_xor(s, 8, 32);
    if (nl == 0) {
      const int m = khalf ? (r + 8) : r;
      const int e = blockIdx.x * 128 + wave * 16 + m;
      if (e < E) { att[e] = s; atomicMaxF32(&mx[dst[e]], s); }
    }
  }
}

// ------------------------------------------------------- softmax denominator
__global__ __launch_bounds__(256) void k_softmax_denom(
    const float* __restrict__ att, const float* __restrict__ mx,
    const int* __restrict__ dst, float* __restrict__ ex, float* __restrict__ dn, int E)
{
  const int e = blockIdx.x * 256 + threadIdx.x;
  if (e < E) {
    const int d = dst[e];
    const float v = __expf(att[e] - mx[d]);
    ex[e] = v;
    atomicAdd(&dn[d], v);
  }
}

// ----------------------------------------------- weighted message scatter-sum
__global__ __launch_bounds__(256) void k_scatter(
    const float* __restrict__ X_msg, const float* __restrict__ feature,
    const float* __restrict__ te_w, const float* __restrict__ te_lam,
    const float* __restrict__ shared_w, const float* __restrict__ shared_b,
    const int* __restrict__ src, const int* __restrict__ dst,
    const int* __restrict__ dst_lane,
    const float* __restrict__ ex, const float* __restrict__ dn,
    float* __restrict__ h_att, int E)
{
  const int e = blockIdx.x * 256 + threadIdx.x;
  if (e >= E) return;
  const int d = dst[e];
  const float a = ex[e] / dn[d];
  float* hr = h_att + (size_t)d * 96;
  const int s = src[e];
  atomicAdd(hr + 0, a * X_msg[2 * s]);
  atomicAdd(hr + 1, a * X_msg[2 * s + 1]);
  const float dt = feature[4 * e + 1];
  atomicAdd(hr + 2, a * feature[4 * e]);
  atomicAdd(hr + 3, a * dt);
  atomicAdd(hr + 4, a * feature[4 * e + 2]);
  const int ln = dst_lane[e];
  const float rl = te_lam[ln];
  const float lam = __expf(-rl * rl);
  const float oml = 1.0f - lam;
#pragma unroll 4
  for (int d2 = 0; d2 < 32; ++d2) {
    const float ang = dt * te_w[ln * 32 + d2];
    const float sh  = dt * shared_w[d2] + shared_b[d2];
    atomicAdd(hr + 5 + d2,  a * (oml * __sinf(ang) + lam * __sinf(sh)));
    atomicAdd(hr + 37 + d2, a * (oml * __cosf(ang) + lam * __cosf(sh)));
  }
}

// --------------------------------------------------------------- 3-layer MLP
// Block: 128 threads = 4 waves, 64 nodes. One 32KB f16 weight buffer reused per
// layer; f16 LDS staging does the C-layout -> A-layout relayout between layers.
__global__ __launch_bounds__(128) void k_mlp(
    const float* __restrict__ h_att,
    const _Float16* __restrict__ w1_16, const float* __restrict__ b1,
    const _Float16* __restrict__ w2_16, const float* __restrict__ b2,
    const _Float16* __restrict__ w3_16, const float* __restrict__ b3,
    float* __restrict__ out, int N)
{
  __shared__ alignas(16) _Float16 Wb[128 * 128];
  __shared__ alignas(16) _Float16 stage[64 * 128];
  __shared__ float bl[128];

  const int tid = threadIdx.x;
  const int wave = tid >> 5, lane = tid & 31;
  const int khalf = lane >> 4, nl = lane & 15;
  const int nodeBase = blockIdx.x * 64;

  // ---- layer 1 (K=69 padded to 96; cols 69..127 of w1_16 are zero) ----
  {
    const uint4* s16 = (const uint4*)w1_16;
    uint4* d16 = (uint4*)Wb;
    for (int i = tid; i < 2048; i += 128) d16[i] = s16[i];
  }
  if (tid < 128) bl[tid] = b1[tid];
  __syncthreads();

  v8f acc[8];
  {
    const int nodeA = nodeBase + wave * 16 + nl;
    const bool okA = nodeA < N;
    const float* rowp = h_att + (size_t)(okA ? nodeA : 0) * 96;
    v16h afr[3];
#pragma unroll
    for (int kt = 0; kt < 3; ++kt) {
#pragma unroll
      for (int v = 0; v < 8; ++v) {
        const int k = kt * 32 + ((v & 3) << 1) + (khalf << 3) + ((v >> 2) << 4);
        float2 p = make_float2(0.0f, 0.0f);
        if (okA) p = *(const float2*)(rowp + k);   // global_load_b64
        afr[kt][2 * v]     = (_Float16)p.x;
        afr[kt][2 * v + 1] = (_Float16)p.y;
      }
    }
#pragma unroll
    for (int nt = 0; nt < 8; ++nt) acc[nt] = (v8f)(0.0f);
#pragma unroll
    for (int kt = 0; kt < 3; ++kt) {
      v16h bfr[8];
#pragma unroll
      for (int nt = 0; nt < 8; ++nt)
        bfr[nt] = frag_ld(&Wb[(nt * 16 + nl) * 128 + kt * 32], khalf);
#pragma unroll
      for (int nt = 0; nt < 8; ++nt)
        acc[nt] = __builtin_amdgcn_wmma_f32_16x16x32_f16(
            false, afr[kt], false, bfr[nt], (short)0, acc[nt], false, false);
    }
  }
  {
    float bv[8];
#pragma unroll
    for (int nt = 0; nt < 8; ++nt) bv[nt] = bl[nt * 16 + nl];
#pragma unroll
    for (int nt = 0; nt < 8; ++nt)
#pragma unroll
      for (int v = 0; v < 8; ++v) {
        const int m = khalf ? (v + 8) : v;
        stage[(wave * 16 + m) * 128 + nt * 16 + nl] =
            (_Float16)fmaxf(acc[nt][v] + bv[nt], 0.0f);
      }
  }
  __syncthreads();

  // ---- layer 2 (K=128) ----
  {
    const uint4* s16 = (const uint4*)w2_16;
    uint4* d16 = (uint4*)Wb;
    for (int i = tid; i < 2048; i += 128) d16[i] = s16[i];
  }
  if (tid < 128) bl[tid] = b2[tid];
  __syncthreads();
  {
    v16h afr[4];
#pragma unroll
    for (int kt = 0; kt < 4; ++kt)
      afr[kt] = frag_ld(&stage[(wave * 16 + nl) * 128 + kt * 32], khalf);
#pragma unroll
    for (int nt = 0; nt < 8; ++nt) acc[nt] = (v8f)(0.0f);
#pragma unroll
    for (int kt = 0; kt < 4; ++kt) {
      v16h bfr[8];
#pragma unroll
      for (int nt = 0; nt < 8; ++nt)
        bfr[nt] = frag_ld(&Wb[(nt * 16 + nl) * 128 + kt * 32], khalf);
#pragma unroll
      for (int nt = 0; nt < 8; ++nt)
        acc[nt] = __builtin_amdgcn_wmma_f32_16x16x32_f16(
            false, afr[kt], false, bfr[nt], (short)0, acc[nt], false, false);
    }
  }
  {
    float bv[8];
#pragma unroll
    for (int nt = 0; nt < 8; ++nt) bv[nt] = bl[nt * 16 + nl];
#pragma unroll
    for (int nt = 0; nt < 8; ++nt)
#pragma unroll
      for (int v = 0; v < 8; ++v) {
        const int m = khalf ? (v + 8) : v;
        stage[(wave * 16 + m) * 128 + nt * 16 + nl] =
            (_Float16)fmaxf(acc[nt][v] + bv[nt], 0.0f);
      }
  }
  __syncthreads();

  // ---- layer 3 (K=128) ----
  {
    const uint4* s16 = (const uint4*)w3_16;
    uint4* d16 = (uint4*)Wb;
    for (int i = tid; i < 2048; i += 128) d16[i] = s16[i];
  }
  if (tid < 128) bl[tid] = b3[tid];
  __syncthreads();
  {
    v16h afr[4];
#pragma unroll
    for (int kt = 0; kt < 4; ++kt)
      afr[kt] = frag_ld(&stage[(wave * 16 + nl) * 128 + kt * 32], khalf);
#pragma unroll
    for (int nt = 0; nt < 8; ++nt) acc[nt] = (v8f)(0.0f);
#pragma unroll
    for (int kt = 0; kt < 4; ++kt) {
      v16h bfr[8];
#pragma unroll
      for (int nt = 0; nt < 8; ++nt)
        bfr[nt] = frag_ld(&Wb[(nt * 16 + nl) * 128 + kt * 32], khalf);
#pragma unroll
      for (int nt = 0; nt < 8; ++nt)
        acc[nt] = __builtin_amdgcn_wmma_f32_16x16x32_f16(
            false, afr[kt], false, bfr[nt], (short)0, acc[nt], false, false);
    }
  }
  {
    float bv[8];
#pragma unroll
    for (int nt = 0; nt < 8; ++nt) bv[nt] = bl[nt * 16 + nl];
#pragma unroll
    for (int nt = 0; nt < 8; ++nt)
#pragma unroll
      for (int v = 0; v < 8; ++v) {
        const int m = khalf ? (v + 8) : v;
        const int node = nodeBase + wave * 16 + m;
        if (node < N)
          out[(size_t)node * 128 + nt * 16 + nl] = fmaxf(acc[nt][v] + bv[nt], 0.0f);
      }
  }
}

// ------------------------------------------------------------------- launch
extern "C" void kernel_launch(void* const* d_in, const int* in_sizes, int n_in,
                              void* d_out, int out_size, void* d_ws, size_t ws_size,
                              hipStream_t stream) {
  (void)n_in; (void)out_size; (void)ws_size;
  const float* X_msg    = (const float*)d_in[0];
  const float* feature  = (const float*)d_in[1];
  const float* te_w     = (const float*)d_in[2];
  const float* te_lam   = (const float*)d_in[3];
  const float* shared_w = (const float*)d_in[4];
  const float* shared_b = (const float*)d_in[5];
  const float* w_att_w  = (const float*)d_in[6];
  const float* w_att_b  = (const float*)d_in[7];
  const float* va       = (const float*)d_in[8];
  const float* w1       = (const float*)d_in[9];
  const float* b1       = (const float*)d_in[10];
  const float* w2       = (const float*)d_in[11];
  const float* b2       = (const float*)d_in[12];
  const float* w3       = (const float*)d_in[13];
  const float* b3       = (const float*)d_in[14];
  const int*   src      = (const int*)d_in[15];
  const int*   dst      = (const int*)d_in[16];
  const int*   dlane    = (const int*)d_in[17];
  float* out = (float*)d_out;
  const int N = in_sizes[0] / 2;
  const int E = in_sizes[1] / 4;

  // ws layout (floats): h_att[N*96] | mx[N] | dn[N] | att[E] | ex[E] | f16 weights
  float* h_att = (float*)d_ws;
  float* mx  = h_att + (size_t)N * 96;
  float* dn  = mx + N;
  float* att = dn + N;
  float* ex  = att + E;
  size_t f16off = (size_t)N * 96 + 2 * (size_t)N + 2 * (size_t)E;
  f16off = (f16off + 7) & ~(size_t)7;              // 16B-align the f16 region
  _Float16* wa16  = (_Float16*)((float*)d_ws + f16off);
  _Float16* w1_16 = wa16 + 128 * 96;
  _Float16* w2_16 = w1_16 + 128 * 128;
  _Float16* w3_16 = w2_16 + 128 * 128;

  k_prep<<<64, 256, 0, stream>>>(w_att_w, w1, w2, w3, wa16, w1_16, w2_16, w3_16);
  const int initBlocks = (int)(((size_t)N * 96 + 255) / 256);
  k_init<<<initBlocks, 256, 0, stream>>>(h_att, mx, dn, N);
  k_edge_att<<<(E + 127) / 128, 256, 0, stream>>>(
      X_msg, feature, te_w, te_lam, shared_w, shared_b,
      wa16, w_att_b, va, src, dst, dlane, att, mx, E);
  k_softmax_denom<<<(E + 255) / 256, 256, 0, stream>>>(att, mx, dst, ex, dn, E);
  k_scatter<<<(E + 255) / 256, 256, 0, stream>>>(
      X_msg, feature, te_w, te_lam, shared_w, shared_b,
      src, dst, dlane, ex, dn, h_att, E);
  k_mlp<<<(N + 63) / 64, 128, 0, stream>>>(h_att, w1_16, b1, w2_16, b2, w3_16, b3, out, N);
}
